// Spline2DBase_6133213298946
// MI455X (gfx1250) — matrix operand order repaired and checked by reading, not compiled, not run on hardware
//
#include <hip/hip_runtime.h>
#include <hip/hip_bf16.h>
#include <math.h>

typedef __attribute__((ext_vector_type(16))) _Float16 v16h;
typedef __attribute__((ext_vector_type(8)))  float    v8f;

#define KBINS 8
#define BOUND 5.0f
#define MIN_BW 0.001f
#define MIN_BH 0.001f
#define MIN_D 0.001f
#define MIN_L 0.025f
#define SPEPS 1e-6f

// ---------- workspace float layout ----------
// [0..7]   w1 bin widths       [8..16]  cw1 knots (9)
// [17..24] h1 bin heights      [25..33] ch1 knots (9)
// [34..42] d1 full (9)         [43..50] lam1 (8)
// [51] scale0  [52] scale1  [53] log(scale0)+log(scale1)
// [56..87] b2 padded to 32 (last = 0)
// halfs at float offset 96: BW1L[512] BW1R[512] BW2L[512] BW2R[512]

__device__ __forceinline__ float softplus_f(float x) {
    return (x > 20.f) ? x : log1pf(expf(x));
}
__device__ __forceinline__ float sigmoid_f(float x) {
    return 1.f / (1.f + expf(-x));
}

// select a[idx] from a register array, idx in [0,7], fully unrolled cndmask chain
__device__ __forceinline__ float sel8(const float* a, int idx) {
    float r = a[0];
#pragma unroll
    for (int i = 1; i < 8; ++i) r = (idx == i) ? a[i] : r;
    return r;
}

// linear rational spline evaluation; all table arrays live in registers.
__device__ __forceinline__ void lrs_eval(float x,
                                         const float* cw, const float* wbin,
                                         const float* ch, const float* hbin,
                                         const float* dfull, const float* lam,
                                         float& y, float& ld) {
    int s = 0;
#pragma unroll
    for (int i = 0; i < 9; ++i) s += (x >= cw[i] + SPEPS) ? 1 : 0;
    int bin = s - 1;
    bin = bin < 0 ? 0 : (bin > 7 ? 7 : bin);

    float iw  = sel8(wbin, bin);
    float icw = sel8(cw, bin);        // cw[bin], bin<=7
    float ich = sel8(ch, bin);
    float ih  = sel8(hbin, bin);
    float da  = sel8(dfull, bin);     // dfull[bin]
    float db  = sel8(dfull + 1, bin); // dfull[bin+1]
    float il  = sel8(lam, bin);

    float idelta = ih / iw;
    float wb = sqrtf(da / db);
    float wc = (il * da + (1.f - il) * wb * db) / idelta;
    float ya = ich;
    float yb = ih + ich;
    float yc = ((1.f - il) * ya + il * wb * yb) / ((1.f - il) + il * wb);
    float theta = (x - icw) / iw;
    bool  low = theta <= il;
    float num  = low ? (ya * (il - theta) + wc * yc * theta)
                     : (wc * yc * (1.f - theta) + wb * yb * (theta - il));
    float den  = low ? ((il - theta) + wc * theta)
                     : (wc * (1.f - theta) + wb * (theta - il));
    float dnum = (low ? (wc * il * (yc - ya))
                      : (wb * wc * (1.f - il) * (yb - yc))) * iw;
    y  = num / den;
    ld = logf(dnum) - 2.f * logf(fabsf(den)) - 2.f * logf(iw);
    bool inside = (x >= -BOUND) && (x <= BOUND);
    if (!inside) { y = x; ld = 0.f; }
}

// build spline tables (widths/knots) from 8 raw logits -> registers or ws
__device__ __forceinline__ void build_knots(const float* raw, float minb,
                                            float* bins, float* knots) {
    float e[8];
    float mx = raw[0];
#pragma unroll
    for (int i = 1; i < 8; ++i) mx = fmaxf(mx, raw[i]);
    float sum = 0.f;
#pragma unroll
    for (int i = 0; i < 8; ++i) { e[i] = expf(raw[i] - mx); sum += e[i]; }
    float inv = 1.f / sum;
    float fac = 1.f - minb * 8.f;
    knots[0] = -BOUND;
    float c = 0.f;
#pragma unroll
    for (int i = 0; i < 8; ++i) {
        float wi = minb + fac * (e[i] * inv);
        c += wi;
        knots[i + 1] = (i == 7) ? BOUND : fmaf(2.f * BOUND, c, -BOUND);
    }
#pragma unroll
    for (int i = 0; i < 8; ++i) bins[i] = knots[i + 1] - knots[i];
}

// ---------------- setup kernel: run once, 1 block -------------------
__global__ void spline_setup_kernel(const float* __restrict__ p1,
                                    const float* __restrict__ W1,
                                    const float* __restrict__ W2,
                                    const float* __restrict__ b2,
                                    const float* __restrict__ uscale,
                                    float* __restrict__ ws) {
    if (threadIdx.x == 0) {
        float wraw[8], hraw[8], bins[8], knots[9];
#pragma unroll
        for (int i = 0; i < 8; ++i) { wraw[i] = p1[i]; hraw[i] = p1[8 + i]; }
        build_knots(wraw, MIN_BW, bins, knots);
#pragma unroll
        for (int i = 0; i < 8; ++i) ws[i] = bins[i];
#pragma unroll
        for (int i = 0; i < 9; ++i) ws[8 + i] = knots[i];
        build_knots(hraw, MIN_BH, bins, knots);
#pragma unroll
        for (int i = 0; i < 8; ++i) ws[17 + i] = bins[i];
#pragma unroll
        for (int i = 0; i < 9; ++i) ws[25 + i] = knots[i];
        ws[34] = 1.f - MIN_D;
        ws[42] = 1.f - MIN_D;
#pragma unroll
        for (int j = 0; j < 7; ++j) ws[35 + j] = MIN_D + softplus_f(p1[16 + j]);
#pragma unroll
        for (int i = 0; i < 8; ++i)
            ws[43 + i] = (1.f - 2.f * MIN_L) * sigmoid_f(p1[23 + i]) + MIN_L;
        float s0 = softplus_f(uscale[0]);
        float s1 = softplus_f(uscale[1]);
        ws[51] = s0; ws[52] = s1; ws[53] = logf(s0) + logf(s1);
    }
    if (threadIdx.x < 32)
        ws[56 + threadIdx.x] = (threadIdx.x < 31) ? b2[threadIdx.x] : 0.f;

    // Repack W1 (32x32) and W2 (32x31, pad to 32) into WMMA B-operand
    // per-lane order: lane l holds column n=l&15, h=l>>4; element j holds
    // K = 8h+j (j<8) or 16+8h+(j-8) (j>=8).
    _Float16* wsH = (_Float16*)(ws + 96);
    for (int e = threadIdx.x; e < 2048; e += blockDim.x) {
        int tile = e >> 9;       // 0:W1 cols0-15, 1:W1 cols16-31, 2:W2 cols0-15, 3:W2 cols16-31
        int idx  = e & 511;
        int l = idx >> 4, j = idx & 15;
        int k = (j < 8) ? ((l & 16) >> 1) + j : 8 + ((l & 16) >> 1) + j;    // K index
        int n = l & 15;          // column within tile
        float v;
        if (tile < 2) {
            v = W1[k * 32 + n + 16 * tile];
        } else {
            int c = n + 16 * (tile - 2);
            v = (c < 31) ? W2[k * 31 + c] : 0.f;
        }
        wsH[e] = (_Float16)v;
    }
}

// ---------------- main kernel -------------------
__global__ __launch_bounds__(256) void spline_main_kernel(
    const float* __restrict__ x,
    const float* __restrict__ W0, const float* __restrict__ b0,
    const float* __restrict__ b1, const float* __restrict__ loc,
    const float* __restrict__ ws, float* __restrict__ out, int n) {

    __shared__ __align__(32) _Float16 ldsA[8][512];   // 8 KB: layer-2 A staging
    __shared__ float ldsP[8][1024];                   // 32 KB: params transpose

    const int lane = threadIdx.x & 31;
    const int wid  = threadIdx.x >> 5;
    const long waveBase = (long)blockIdx.x * 256 + (long)wid * 32;
    const int nlo = lane & 15;
    const int khi = (lane & 16) ? 8 : 0;   // A-operand K-half select
    const int mbase = (lane & 16) ? 8 : 0; // D-layout row-half

    const _Float16* wsH = (const _Float16*)(ws + 96);
    v16h Bw1l = ((const v16h*)(wsH))[lane];
    v16h Bw1r = ((const v16h*)(wsH + 512))[lane];
    v16h Bw2l = ((const v16h*)(wsH + 1024))[lane];
    v16h Bw2r = ((const v16h*)(wsH + 1536))[lane];

    const float b1l = b1[nlo], b1r = b1[nlo + 16];
    const float b2l = ws[56 + nlo], b2r = ws[56 + 16 + nlo];

    // layer-0 weights for this lane's 16 K slots (shared by both M-tiles)
    float w0v[16], b0v[16];
#pragma unroll
    for (int j = 0; j < 16; ++j) {
        int k = (j < 8) ? (j + khi) : (j + 8 + khi);
        w0v[j] = W0[k];
        b0v[j] = b0[k];
    }

    const int holderBase = (nlo < 8) ? 0 : 16;
    const int jl = nlo & 7;

#pragma unroll
    for (int t = 0; t < 2; ++t) {
        // ---- layer 0: build A operand directly (rank-1 layer, no LDS) ----
        float x1t = x[(waveBase + t * 16 + nlo) * 2];
        v16h A1;
#pragma unroll
        for (int j = 0; j < 16; ++j) {
            float hv = fmaxf(fmaf(x1t, w0v[j], b0v[j]), 0.f);
            A1[j] = (_Float16)hv;
        }

        // ---- layer 1: two 16x16x32 f16 WMMAs, bias via C, then ReLU ----
        v8f cl, cr;
#pragma unroll
        for (int i = 0; i < 8; ++i) { cl[i] = b1l; cr[i] = b1r; }
        v8f h1l = __builtin_amdgcn_wmma_f32_16x16x32_f16(false, A1, false, Bw1l,
                                                         (short)0, cl, false, false);
        v8f h1r = __builtin_amdgcn_wmma_f32_16x16x32_f16(false, A1, false, Bw1r,
                                                         (short)0, cr, false, false);
#pragma unroll
        for (int i = 0; i < 8; ++i) {
            h1l[i] = fmaxf(h1l[i], 0.f);
            h1r[i] = fmaxf(h1r[i], 0.f);
        }

        // ---- transpose D-layout -> A-layout through LDS (f16) ----
#pragma unroll
        for (int r = 0; r < 8; ++r) {
            int m = r + mbase;
            ldsA[wid][(m + holderBase) * 16 + jl]     = (_Float16)h1l[r]; // feature nlo
            ldsA[wid][(m + holderBase) * 16 + jl + 8] = (_Float16)h1r[r]; // feature nlo+16
        }
        __syncthreads();
        v16h A2 = *(const v16h*)&ldsA[wid][lane * 16];

        // ---- layer 2: two WMMAs (W2 padded to 32 cols), bias via C ----
        v8f c2l, c2r;
#pragma unroll
        for (int i = 0; i < 8; ++i) { c2l[i] = b2l; c2r[i] = b2r; }
        v8f pL = __builtin_amdgcn_wmma_f32_16x16x32_f16(false, A2, false, Bw2l,
                                                        (short)0, c2l, false, false);
        v8f pR = __builtin_amdgcn_wmma_f32_16x16x32_f16(false, A2, false, Bw2r,
                                                        (short)0, c2r, false, false);

        // ---- transpose params back: lane owns column nlo, rows mbase..mbase+7 ----
#pragma unroll
        for (int r = 0; r < 8; ++r) {
            int p = t * 16 + r + mbase;
            ldsP[wid][p * 32 + nlo]      = pL[r];
            ldsP[wid][p * 32 + 16 + nlo] = pR[r];
        }
        __syncthreads();  // also protects ldsA reuse across tiles (WAR)
    }

    // ---------------- per-point stage ----------------
    long p = waveBase + lane;
    float x1 = x[p * 2];
    float x2 = x[p * 2 + 1];

    // spline1: load precomputed tables (wave-uniform addresses -> scalar loads)
    float cw1[9], wb1[8], ch1[9], hb1[8], df1[9], lm1[8];
#pragma unroll
    for (int i = 0; i < 8; ++i) wb1[i] = ws[i];
#pragma unroll
    for (int i = 0; i < 9; ++i) cw1[i] = ws[8 + i];
#pragma unroll
    for (int i = 0; i < 8; ++i) hb1[i] = ws[17 + i];
#pragma unroll
    for (int i = 0; i < 9; ++i) ch1[i] = ws[25 + i];
#pragma unroll
    for (int i = 0; i < 9; ++i) df1[i] = ws[34 + i];
#pragma unroll
    for (int i = 0; i < 8; ++i) lm1[i] = ws[43 + i];
    float y1, ld1;
    lrs_eval(x1, cw1, wb1, ch1, hb1, df1, lm1, y1, ld1);

    // spline2: this lane's 31 MLP params from LDS
    float pr[31];
#pragma unroll
    for (int i = 0; i < 31; ++i) pr[i] = ldsP[wid][lane * 32 + i];

    float wb2[8], cw2[9], hb2[8], ch2[9], df2[9], lm2[8];
    build_knots(pr, MIN_BW, wb2, cw2);
    build_knots(pr + 8, MIN_BH, hb2, ch2);
    df2[0] = 1.f - MIN_D;
    df2[8] = 1.f - MIN_D;
#pragma unroll
    for (int j = 0; j < 7; ++j) df2[1 + j] = MIN_D + softplus_f(pr[16 + j]);
#pragma unroll
    for (int i = 0; i < 8; ++i)
        lm2[i] = (1.f - 2.f * MIN_L) * sigmoid_f(pr[23 + i]) + MIN_L;
    float y2, ld2;
    lrs_eval(x2, cw2, wb2, ch2, hb2, df2, lm2, y2, ld2);

    float s0 = ws[51], s1 = ws[52], lss = ws[53];
    out[p * 2]     = loc[0] + s0 * y1;
    out[p * 2 + 1] = loc[1] + s1 * y2;
    out[(long)2 * n + p] = ld1 + ld2 + lss;
}

extern "C" void kernel_launch(void* const* d_in, const int* in_sizes, int n_in,
                              void* d_out, int out_size, void* d_ws, size_t ws_size,
                              hipStream_t stream) {
    const float* x      = (const float*)d_in[0];
    const float* p1     = (const float*)d_in[1];
    const float* W0     = (const float*)d_in[2];
    const float* b0     = (const float*)d_in[3];
    const float* W1     = (const float*)d_in[4];
    const float* b1     = (const float*)d_in[5];
    const float* W2     = (const float*)d_in[6];
    const float* b2     = (const float*)d_in[7];
    const float* loc    = (const float*)d_in[8];
    const float* uscale = (const float*)d_in[9];
    float* ws  = (float*)d_ws;
    float* out = (float*)d_out;
    int n = in_sizes[0] / 2;

    spline_setup_kernel<<<1, 64, 0, stream>>>(p1, W1, W2, b2, uscale, ws);
    spline_main_kernel<<<n / 256, 256, 0, stream>>>(x, W0, b0, b1, loc, ws, out, n);
}
